// GATModel_75737453298300
// MI455X (gfx1250) — matrix operand order repaired and checked by reading, not compiled, not run on hardware
//
#include <hip/hip_runtime.h>
#include <hip/hip_bf16.h>

// ---------------------------------------------------------------------------
// Types for CDNA5 WMMA
// ---------------------------------------------------------------------------
typedef __attribute__((ext_vector_type(16))) __bf16 v16bf;
typedef __attribute__((ext_vector_type(8)))  float  v8f;

union FragBF {
    unsigned short us[16];
    uint4          u4[2];
    v16bf          v;
};

__device__ __forceinline__ unsigned short f2bf(float f) {
    unsigned u = __float_as_uint(f);
    unsigned r = (u + 0x7FFFu + ((u >> 16) & 1u)) >> 16;   // round-to-nearest-even
    return (unsigned short)r;
}

// Ordered-uint encoding so unsigned atomicMax == float max (handles negatives)
__device__ __forceinline__ unsigned enc_ord(float f) {
    unsigned u = __float_as_uint(f);
    return (u & 0x80000000u) ? ~u : (u | 0x80000000u);
}
__device__ __forceinline__ float dec_ord(unsigned u) {
    return (u & 0x80000000u) ? __uint_as_float(u ^ 0x80000000u)
                             : __uint_as_float(~u);
}
#define ENC_NEG_INF 0x007FFFFFu   // enc_ord(-inf)

// ---------------------------------------------------------------------------
// WMMA GEMM: C[M x NCOLS] = A[M x 128] @ B[128 x NCOLS]   (fp32 in/out, bf16 MACs)
// Block = 128 threads = 4 waves; wave w owns rows [blk*64 + w*16, +16),
// all NCOLS/16 column tiles. B staged in LDS transposed (K contiguous per col).
// ---------------------------------------------------------------------------
template <int NCOLS>
__global__ void __launch_bounds__(128)
gat_gemm_bf16_k(const float* __restrict__ A, const float* __restrict__ B,
                float* __restrict__ C, int M)
{
    constexpr int K  = 128;
    constexpr int NT = NCOLS / 16;
    __shared__ unsigned short sB[NCOLS * K];   // [col][k], bf16

    const int tid = threadIdx.x;
    for (int i = tid; i < NCOLS * K; i += blockDim.x) {
        int col = i / K, k = i % K;
        sB[i] = f2bf(B[(size_t)k * NCOLS + col]);
    }
    __syncthreads();

    const int wave = tid >> 5;
    const int lane = tid & 31;
    const int half = lane >> 4;       // 0: lanes 0-15, 1: lanes 16-31
    const int l16  = lane & 15;

    const int rowbase = blockIdx.x * 64 + wave * 16;
    int arow = rowbase + l16;
    if (arow >= M) arow = M - 1;      // clamp (EXEC stays all-ones for WMMA)

    v8f acc[NT] = {};                 // zero accumulators

    const float* aptr = A + (size_t)arow * K + half * 8;
    #pragma unroll
    for (int k0 = 0; k0 < K; k0 += 32) {
        // A fragment: 16-bit A 16x32 layout -> lane holds K = k0+half*8+[0..7]
        // and K = k0+16+half*8+[0..7]; both contiguous float4 pairs.
        FragBF aF;
        const float* p = aptr + k0;
        float4 f0 = *(const float4*)(p);
        float4 f1 = *(const float4*)(p + 4);
        float4 f2 = *(const float4*)(p + 16);
        float4 f3 = *(const float4*)(p + 20);
        aF.us[0] = f2bf(f0.x); aF.us[1] = f2bf(f0.y); aF.us[2]  = f2bf(f0.z); aF.us[3]  = f2bf(f0.w);
        aF.us[4] = f2bf(f1.x); aF.us[5] = f2bf(f1.y); aF.us[6]  = f2bf(f1.z); aF.us[7]  = f2bf(f1.w);
        aF.us[8] = f2bf(f2.x); aF.us[9] = f2bf(f2.y); aF.us[10] = f2bf(f2.z); aF.us[11] = f2bf(f2.w);
        aF.us[12]= f2bf(f3.x); aF.us[13]= f2bf(f3.y); aF.us[14] = f2bf(f3.z); aF.us[15] = f2bf(f3.w);

        #pragma unroll
        for (int t = 0; t < NT; t++) {
            // B fragment: lane holds col = t*16+l16, K = k0 + half*8 + [0..7] and k0 + 16 + half*8 + [0..7]
            FragBF bF;
            const unsigned short* bp = &sB[(t * 16 + l16) * K + k0 + half * 8];
            bF.u4[0] = *(const uint4*)(bp);
            bF.u4[1] = *(const uint4*)(bp + 16);
            acc[t] = __builtin_amdgcn_wmma_f32_16x16x32_bf16(
                         false, aF.v, false, bF.v, (short)0, acc[t], false, false);
        }
    }

    // C/D layout: VGPR r, lanes 0-15 -> row r, lanes 16-31 -> row 8+r; col = l16
    const int rrow0 = rowbase + half * 8;
    #pragma unroll
    for (int t = 0; t < NT; t++) {
        const int col = t * 16 + l16;
        #pragma unroll
        for (int r = 0; r < 8; r++) {
            int row = rrow0 + r;
            if (row < M) C[(size_t)row * NCOLS + col] = acc[t][r];
        }
    }
}

// ---------------------------------------------------------------------------
// alpha_s / alpha_d per (node, head):  dot(h[n,hd,:], a_src/dst[hd,:])
// ---------------------------------------------------------------------------
template <int H, int C>
__global__ void gat_alpha_k(const float* __restrict__ h,
                            const float* __restrict__ a_src,
                            const float* __restrict__ a_dst,
                            float* __restrict__ as, float* __restrict__ ad, int n)
{
    long long i = (long long)blockIdx.x * blockDim.x + threadIdx.x;
    if (i >= (long long)n * H) return;
    int node = (int)(i / H), hd = (int)(i % H);
    const float* hp = h + (size_t)node * (H * C) + hd * C;
    float s = 0.f, d = 0.f;
    #pragma unroll
    for (int c = 0; c < C; c++) {
        float v = hp[c];
        s += v * a_src[hd * C + c];
        d += v * a_dst[hd * C + c];
    }
    as[i] = s; ad[i] = d;
}

// ---------------------------------------------------------------------------
// Edge softmax: segment max, then segment sum of exp
// ---------------------------------------------------------------------------
__global__ void gat_fill_u32_k(unsigned* p, unsigned v, long long n) {
    long long i = (long long)blockIdx.x * blockDim.x + threadIdx.x;
    if (i < n) p[i] = v;
}

template <int H>
__global__ void gat_edge_max_k(const int* __restrict__ src, const int* __restrict__ dst,
                               int Etot, int Eorig,
                               const float* __restrict__ as, const float* __restrict__ ad,
                               unsigned* __restrict__ emax)
{
    long long i = (long long)blockIdx.x * blockDim.x + threadIdx.x;
    if (i >= (long long)Etot * H) return;
    int e = (int)(i / H), h = (int)(i % H);
    int s = e < Eorig ? src[e] : (e - Eorig);
    int d = e < Eorig ? dst[e] : (e - Eorig);
    float v = as[(size_t)s * H + h] + ad[(size_t)d * H + h];
    v = v > 0.f ? v : 0.2f * v;                        // leaky_relu(0.2)
    atomicMax(emax + (size_t)d * H + h, enc_ord(v));
}

template <int H>
__global__ void gat_edge_sum_k(const int* __restrict__ src, const int* __restrict__ dst,
                               int Etot, int Eorig,
                               const float* __restrict__ as, const float* __restrict__ ad,
                               const unsigned* __restrict__ emax, float* __restrict__ den)
{
    long long i = (long long)blockIdx.x * blockDim.x + threadIdx.x;
    if (i >= (long long)Etot * H) return;
    int e = (int)(i / H), h = (int)(i % H);
    int s = e < Eorig ? src[e] : (e - Eorig);
    int d = e < Eorig ? dst[e] : (e - Eorig);
    float v = as[(size_t)s * H + h] + ad[(size_t)d * H + h];
    v = v > 0.f ? v : 0.2f * v;
    float ex = __expf(v - dec_ord(emax[(size_t)d * H + h]));
    atomicAdd(den + (size_t)d * H + h, ex);
}

// ---------------------------------------------------------------------------
// Aggregation layer 1: one wave32 per edge; lane j handles channels j, j+32,
// j+64, j+96 (heads 0..3). Softmax weights recomputed (cheaper than storing).
// ---------------------------------------------------------------------------
__global__ void gat_agg1_k(const int* __restrict__ src, const int* __restrict__ dst,
                           int Etot, int Eorig,
                           const float* __restrict__ as, const float* __restrict__ ad,
                           const unsigned* __restrict__ emax, const float* __restrict__ den,
                           const float* __restrict__ h, float* __restrict__ out)
{
    const int lane = threadIdx.x & 31;
    long long e = (long long)blockIdx.x * (blockDim.x >> 5) + (threadIdx.x >> 5);
    if (e >= Etot) return;
    int s = e < Eorig ? src[e] : (int)(e - Eorig);
    int d = e < Eorig ? dst[e] : (int)(e - Eorig);
    float w = 0.f;
    if (lane < 4) {
        float v = as[(size_t)s * 4 + lane] + ad[(size_t)d * 4 + lane];
        v = v > 0.f ? v : 0.2f * v;
        float ex = __expf(v - dec_ord(emax[(size_t)d * 4 + lane]));
        w = ex / (den[(size_t)d * 4 + lane] + 1e-16f);
    }
    float w0 = __shfl(w, 0, 32), w1 = __shfl(w, 1, 32);
    float w2 = __shfl(w, 2, 32), w3 = __shfl(w, 3, 32);
    const float* hs = h + (size_t)s * 128;
    float* od = out + (size_t)d * 128;
    atomicAdd(od + lane,      hs[lane]      * w0);
    atomicAdd(od + lane + 32, hs[lane + 32] * w1);
    atomicAdd(od + lane + 64, hs[lane + 64] * w2);
    atomicAdd(od + lane + 96, hs[lane + 96] * w3);
}

// Aggregation layer 2 (H=1, C=32): one wave per edge, lane per channel
__global__ void gat_agg2_k(const int* __restrict__ src, const int* __restrict__ dst,
                           int Etot, int Eorig,
                           const float* __restrict__ as, const float* __restrict__ ad,
                           const unsigned* __restrict__ emax, const float* __restrict__ den,
                           const float* __restrict__ h, float* __restrict__ out)
{
    const int lane = threadIdx.x & 31;
    long long e = (long long)blockIdx.x * (blockDim.x >> 5) + (threadIdx.x >> 5);
    if (e >= Etot) return;
    int s = e < Eorig ? src[e] : (int)(e - Eorig);
    int d = e < Eorig ? dst[e] : (int)(e - Eorig);
    float w = 0.f;
    if (lane == 0) {
        float v = as[s] + ad[d];
        v = v > 0.f ? v : 0.2f * v;
        w = __expf(v - dec_ord(emax[d])) / (den[d] + 1e-16f);
    }
    w = __shfl(w, 0, 32);
    atomicAdd(out + (size_t)d * 32 + lane, h[(size_t)s * 32 + lane] * w);
}

// out = elu(out + bias)   (elementwise, width = #columns)
__global__ void gat_bias_elu_k(float* __restrict__ x, const float* __restrict__ b,
                               int width, long long total)
{
    long long i = (long long)blockIdx.x * blockDim.x + threadIdx.x;
    if (i >= total) return;
    float v = x[i] + b[(int)(i % width)];
    x[i] = v > 0.f ? v : expm1f(v);
}

// classifier: [N,32] @ [32,2] + bc
__global__ void gat_cls_k(const float* __restrict__ h, const float* __restrict__ Wc,
                          const float* __restrict__ bc, float* __restrict__ out, int n)
{
    int i = blockIdx.x * blockDim.x + threadIdx.x;
    if (i >= n) return;
    float o0 = bc[0], o1 = bc[1];
    const float* hp = h + (size_t)i * 32;
    #pragma unroll
    for (int c = 0; c < 32; c++) {
        float v = hp[c];
        o0 += v * Wc[c * 2 + 0];
        o1 += v * Wc[c * 2 + 1];
    }
    out[(size_t)i * 2 + 0] = o0;
    out[(size_t)i * 2 + 1] = o1;
}

// ---------------------------------------------------------------------------
// Host launcher
// ---------------------------------------------------------------------------
static inline unsigned cdiv_ll(long long a, long long b) { return (unsigned)((a + b - 1) / b); }

extern "C" void kernel_launch(void* const* d_in, const int* in_sizes, int n_in,
                              void* d_out, int out_size, void* d_ws, size_t ws_size,
                              hipStream_t stream)
{
    const float* x    = (const float*)d_in[0];
    const int*   ei   = (const int*)  d_in[1];
    const float* W1   = (const float*)d_in[2];
    const float* aS1  = (const float*)d_in[3];
    const float* aD1  = (const float*)d_in[4];
    const float* b1   = (const float*)d_in[5];
    const float* W2   = (const float*)d_in[6];
    const float* aS2  = (const float*)d_in[7];
    const float* aD2  = (const float*)d_in[8];
    const float* b2   = (const float*)d_in[9];
    const float* Wc   = (const float*)d_in[10];
    const float* bc   = (const float*)d_in[11];

    const int N    = in_sizes[0] / 128;   // F_IN = 128
    const int E    = in_sizes[1] / 2;
    const int Etot = E + N;               // + self-loops
    const int* src = ei;
    const int* dst = ei + E;

    // workspace layout (floats)
    size_t need = ((size_t)N * 256 + (size_t)N * 16) * sizeof(float);
    if (ws_size < need) return;
    float*    h1    = (float*)d_ws;                       // N*128
    float*    out1  = h1   + (size_t)N * 128;             // N*128
    float*    as1   = out1 + (size_t)N * 128;             // N*4
    float*    ad1   = as1  + (size_t)N * 4;               // N*4
    unsigned* emax1 = (unsigned*)(ad1 + (size_t)N * 4);   // N*4
    float*    den1  = (float*)(emax1 + (size_t)N * 4);    // N*4
    // layer-2 scratch aliases h1 region (h1 dead after agg1)
    float*    h2    = h1;                                 // N*32
    float*    out2  = h1 + (size_t)N * 32;                // N*32
    float*    as2   = h1 + (size_t)N * 64;                // N
    float*    ad2   = as2 + N;                            // N
    unsigned* emax2 = (unsigned*)(ad2 + N);               // N
    float*    den2  = (float*)(emax2 + N);                // N

    const unsigned gemm_blocks = cdiv_ll(N, 64);

    // ---------------- layer 1 ----------------
    gat_gemm_bf16_k<128><<<gemm_blocks, 128, 0, stream>>>(x, W1, h1, N);
    gat_alpha_k<4, 32><<<cdiv_ll((long long)N * 4, 256), 256, 0, stream>>>(h1, aS1, aD1, as1, ad1, N);

    gat_fill_u32_k<<<cdiv_ll((long long)N * 4, 256), 256, 0, stream>>>(emax1, ENC_NEG_INF, (long long)N * 4);
    hipMemsetAsync(den1, 0, (size_t)N * 4 * sizeof(float), stream);
    hipMemsetAsync(out1, 0, (size_t)N * 128 * sizeof(float), stream);

    gat_edge_max_k<4><<<cdiv_ll((long long)Etot * 4, 256), 256, 0, stream>>>(src, dst, Etot, E, as1, ad1, emax1);
    gat_edge_sum_k<4><<<cdiv_ll((long long)Etot * 4, 256), 256, 0, stream>>>(src, dst, Etot, E, as1, ad1, emax1, den1);
    gat_agg1_k<<<cdiv_ll(Etot, 8), 256, 0, stream>>>(src, dst, Etot, E, as1, ad1, emax1, den1, h1, out1);
    gat_bias_elu_k<<<cdiv_ll((long long)N * 128, 256), 256, 0, stream>>>(out1, b1, 128, (long long)N * 128);

    // ---------------- layer 2 ----------------
    gat_gemm_bf16_k<32><<<gemm_blocks, 128, 0, stream>>>(out1, W2, h2, N);
    gat_alpha_k<1, 32><<<cdiv_ll(N, 256), 256, 0, stream>>>(h2, aS2, aD2, as2, ad2, N);

    gat_fill_u32_k<<<cdiv_ll(N, 256), 256, 0, stream>>>(emax2, ENC_NEG_INF, N);
    hipMemsetAsync(den2, 0, (size_t)N * sizeof(float), stream);
    hipMemsetAsync(out2, 0, (size_t)N * 32 * sizeof(float), stream);

    gat_edge_max_k<1><<<cdiv_ll(Etot, 256), 256, 0, stream>>>(src, dst, Etot, E, as2, ad2, emax2);
    gat_edge_sum_k<1><<<cdiv_ll(Etot, 256), 256, 0, stream>>>(src, dst, Etot, E, as2, ad2, emax2, den2);
    gat_agg2_k<<<cdiv_ll(Etot, 8), 256, 0, stream>>>(src, dst, Etot, E, as2, ad2, emax2, den2, h2, out2);
    gat_bias_elu_k<<<cdiv_ll((long long)N * 32, 256), 256, 0, stream>>>(out2, b2, 32, (long long)N * 32);

    // ---------------- classifier ----------------
    gat_cls_k<<<cdiv_ll(N, 256), 256, 0, stream>>>(out2, Wc, bc, (float*)d_out, N);
}
